// MantaCachedConvolutionPooling_59785944760883
// MI455X (gfx1250) — compile-verified, matrix operand order checked by reading, not run on hardware
//
#include <hip/hip_runtime.h>
#include <hip/hip_bf16.h>

#define B_   4
#define S_   2048
#define H_   256
#define OUT_ 512

typedef __bf16 bf16_t;
typedef __attribute__((ext_vector_type(16))) __bf16 v16bf;
typedef __attribute__((ext_vector_type(8)))  __bf16 v8bf;
typedef __attribute__((ext_vector_type(8)))  float  v8f;

__device__ __forceinline__ bf16_t f2bf(float f) {
    union { float f; unsigned u; } a; a.f = f;
    unsigned r = a.u + 0x7FFFu + ((a.u >> 16) & 1u);   // round-to-nearest-even
    union { unsigned short u; bf16_t b; } c;
    c.u = (unsigned short)(r >> 16);
    return c.b;
}

// ---------------------------------------------------------------------------
// Kernel 1: per-batch inclusive prefix sums of p and p(1-p)  ->  mean, std
// one block per batch, 256 threads * 8 elements
// ---------------------------------------------------------------------------
__global__ void scan_kernel(const float* __restrict__ probs,
                            float* __restrict__ mean, float* __restrict__ stdv) {
    int b = blockIdx.x, tid = threadIdx.x;
    const float* P = probs + (size_t)b * S_;
    int t0 = tid * 8;
    float pv[8], qv[8], sp = 0.f, sq = 0.f;
    for (int i = 0; i < 8; ++i) {
        float p = P[t0 + i];
        if (t0 + i == 0) p = 0.f;
        float q = p * (1.f - p);
        sp += p; sq += q;
        pv[i] = sp; qv[i] = sq;
    }
    __shared__ float spS[256], sqS[256];
    spS[tid] = sp; sqS[tid] = sq;
    __syncthreads();
    for (int off = 1; off < 256; off <<= 1) {
        float a = spS[tid], c = sqS[tid];
        float a2 = 0.f, c2 = 0.f;
        if (tid >= off) { a2 = spS[tid - off]; c2 = sqS[tid - off]; }
        __syncthreads();
        spS[tid] = a + a2; sqS[tid] = c + c2;
        __syncthreads();
    }
    float offp = (tid > 0) ? spS[tid - 1] : 0.f;
    float offq = (tid > 0) ? sqS[tid - 1] : 0.f;
    float* M  = mean + (size_t)b * S_;
    float* Sd = stdv + (size_t)b * S_;
    for (int i = 0; i < 8; ++i) {
        M[t0 + i]  = offp + pv[i];
        Sd[t0 + i] = sqrtf(offq + qv[i] + 5e-6f);
    }
}

// ---------------------------------------------------------------------------
// Kernel 2: softmax denominator per (b,t), windowed in j (Gaussian tail
// underflows past |m-j| > sqrt(176*s))
// ---------------------------------------------------------------------------
__global__ void denom_kernel(const float* __restrict__ mean,
                             const float* __restrict__ stdv,
                             float* __restrict__ invd, int J) {
    int idx = blockIdx.x * 256 + threadIdx.x;
    if (idx >= B_ * S_) return;
    float m = mean[idx], s = stdv[idx];
    float w = sqrtf(176.f * s) + 2.f;
    int lo = max(0, (int)floorf(m - w));
    int hi = min(J - 1, (int)ceilf(m + w));
    float inv2s = 1.f / (2.f * s), acc = 0.f;
    for (int j = lo; j <= hi; ++j) {
        float d = m - (float)j;
        acc += __expf(-d * d * inv2s);
    }
    invd[idx] = 1.f / fmaxf(acc, 1e-30f);
}

// ---------------------------------------------------------------------------
// Kernel 3: conv (K=3, per-channel) + transpose -> convT[b][c][t] in bf16
// 32x32 tile transpose through LDS (pitch 33 -> conflict free)
// ---------------------------------------------------------------------------
__global__ void conv_kernel(const float* __restrict__ emb,
                            const float* __restrict__ cw,
                            bf16_t* __restrict__ convT) {
    __shared__ float tile[32][33];
    int b   = blockIdx.x >> 6;          // S_/32 = 64 tiles along t
    int tlo = (blockIdx.x & 63) * 32;
    int clo = blockIdx.y * 32;
    int tx = threadIdx.x, ty = threadIdx.y;  // 32 x 8
    for (int r = 0; r < 4; ++r) {
        int t = tlo + ty + r * 8;
        int c = clo + tx;
        float acc = 0.f;
        #pragma unroll
        for (int i = 0; i < 3; ++i) {
            int ti = t + i - 1;
            if (ti >= 0 && ti < S_)
                acc += cw[i * H_ + c] * emb[((size_t)b * S_ + ti) * H_ + c];
        }
        tile[ty + r * 8][tx] = acc;
    }
    __syncthreads();
    for (int r = 0; r < 4; ++r) {
        int cl = ty + r * 8;   // channel within tile
        int tl = tx;           // t within tile
        convT[((size_t)b * H_ + clo + cl) * S_ + tlo + tl] = f2bf(tile[tl][cl]);
    }
}

// ---------------------------------------------------------------------------
// Kernel 4: one block per (b,j). Writes block_byte_proba, attention_mask,
// regularized_token_size, and (for j<256) the bf16 proba_norm GEMM-A operand.
// ---------------------------------------------------------------------------
__global__ void bbp_kernel(const float* __restrict__ mean,
                           const float* __restrict__ stdv,
                           const float* __restrict__ invd,
                           float* __restrict__ bbp, float* __restrict__ mask,
                           float* __restrict__ reg_out, bf16_t* __restrict__ Abf,
                           int J) {
    int b = blockIdx.x / J, j = blockIdx.x % J;
    int tid = threadIdx.x;
    size_t col  = (size_t)b * S_;
    size_t base = ((size_t)b * J + j) * S_;
    float fj = (float)j;
    float vloc[8], sum = 0.f;
    for (int i = 0; i < 8; ++i) {
        int t = i * 256 + tid;
        float m = mean[col + t], s = stdv[col + t];
        float d = m - fj;
        float v = __expf(-d * d / (2.f * s)) * invd[col + t];
        vloc[i] = v; sum += v;
        bbp[base + t]  = v;
        mask[base + t] = 1.f;
    }
    __shared__ float red[256];
    __shared__ float regS;
    red[tid] = sum; __syncthreads();
    for (int o = 128; o > 0; o >>= 1) {
        if (tid < o) red[tid] += red[tid + o];
        __syncthreads();
    }
    if (tid == 0) {
        float r = fmaxf(red[0], 1.f);
        reg_out[(size_t)b * J + j] = r;
        regS = r;
    }
    __syncthreads();
    if (j < 256) {
        float rinv = 1.f / regS;
        size_t abase = ((size_t)b * 256 + j) * S_;
        for (int i = 0; i < 8; ++i)
            Abf[abase + i * 256 + tid] = f2bf(vloc[i] * rinv);
    }
}

// ---------------------------------------------------------------------------
// Weight transpose + bf16 convert:  Wt[n][k] = (bf16) W[k][n]
// ---------------------------------------------------------------------------
__global__ void wt_kernel(const float* __restrict__ W, bf16_t* __restrict__ Wt,
                          int K, int N) {
    int idx = blockIdx.x * 256 + threadIdx.x;
    if (idx >= N * K) return;
    int n = idx / K, k = idx % K;
    Wt[idx] = f2bf(W[(size_t)k * N + n]);
}

// ---------------------------------------------------------------------------
// WMMA core: one wave computes a 16x16 f32 tile of A(MxK) * Bt(NxK)^T.
// bf16 operands; lane layout matches 16-bit A/B operand maps in the CDNA5 ISA:
//   lanes 0-15 hold row r, K halves {sub*8..sub*8+7} and {+16}; sub=lane>>4.
// ---------------------------------------------------------------------------
__device__ __forceinline__ v8f wmma_tile16(const bf16_t* __restrict__ A,
                                           const bf16_t* __restrict__ Bt,
                                           int K, int lane) {
    int r = lane & 15, sub = lane >> 4;
    const bf16_t* arow = A  + (size_t)r * K;
    const bf16_t* brow = Bt + (size_t)r * K;
    v8f acc = {};
    for (int k = 0; k < K; k += 32) {
        v8bf a0 = *(const v8bf*)(arow + k + sub * 8);
        v8bf a1 = *(const v8bf*)(arow + k + 16 + sub * 8);
        v8bf b0 = *(const v8bf*)(brow + k + sub * 8);
        v8bf b1 = *(const v8bf*)(brow + k + 16 + sub * 8);
        v16bf av = __builtin_shufflevector(a0, a1, 0,1,2,3,4,5,6,7,8,9,10,11,12,13,14,15);
        v16bf bv = __builtin_shufflevector(b0, b1, 0,1,2,3,4,5,6,7,8,9,10,11,12,13,14,15);
        acc = __builtin_amdgcn_wmma_f32_16x16x32_bf16(false, av, false, bv,
                                                      (short)0, acc, false, false);
    }
    return acc;
}

// Generic GEMM: C = A * Bt^T (+bias), optional bf16 mirror of C.
// grid.y = batch, 8 waves/block, one 16x16 tile per wave.
__global__ void gemm_bf16_kernel(const bf16_t* __restrict__ A,
                                 const bf16_t* __restrict__ Bt,
                                 float* __restrict__ C, bf16_t* __restrict__ Cb,
                                 const float* __restrict__ bias,
                                 int M, int N, int K,
                                 long long sA, long long sB, long long sC) {
    long long batch = blockIdx.y;
    A += batch * sA; Bt += batch * sB; C += batch * sC;
    if (Cb) Cb += batch * sC;
    int wave = threadIdx.x >> 5, lane = threadIdx.x & 31;
    int tile = blockIdx.x * 8 + wave;
    int tN = N >> 4;
    if (tile >= (M >> 4) * tN) return;
    int rt = tile / tN, ct = tile % tN;
    v8f acc = wmma_tile16(A + (size_t)rt * 16 * K, Bt + (size_t)ct * 16 * K, K, lane);
    int r = lane & 15, sub = lane >> 4;
    int colg = ct * 16 + r;
    float bv = bias ? bias[colg] : 0.f;
    #pragma unroll
    for (int i = 0; i < 8; ++i) {
        int rowg = rt * 16 + sub * 8 + i;
        float v = acc[i] + bv;
        C[(size_t)rowg * N + colg] = v;
        if (Cb) Cb[(size_t)rowg * N + colg] = f2bf(v);
    }
}

// Output GEMM: rows are (b*256 + j); only j < pad is stored, packed per batch.
__global__ void gemm_out_kernel(const bf16_t* __restrict__ A,
                                const bf16_t* __restrict__ Bt,
                                float* __restrict__ out,
                                const float* __restrict__ bias,
                                int M, int N, int K, int pad) {
    int wave = threadIdx.x >> 5, lane = threadIdx.x & 31;
    int tile = blockIdx.x * 8 + wave;
    int tN = N >> 4;
    if (tile >= (M >> 4) * tN) return;
    int rt = tile / tN, ct = tile % tN;
    v8f acc = wmma_tile16(A + (size_t)rt * 16 * K, Bt + (size_t)ct * 16 * K, K, lane);
    int r = lane & 15, sub = lane >> 4;
    int colg = ct * 16 + r;
    float bv = bias[colg];
    #pragma unroll
    for (int i = 0; i < 8; ++i) {
        int rowg = rt * 16 + sub * 8 + i;
        int b = rowg >> 8, j = rowg & 255;
        if (j < pad)
            out[((size_t)b * pad + j) * N + colg] = acc[i] + bv;
    }
}

// ---------------------------------------------------------------------------
// Highway elementwise: x = sig(gate)*x + (1-sig(gate))*relu(nonlinear)
// ---------------------------------------------------------------------------
__global__ void hw_elem_kernel(const float* __restrict__ proj,
                               float* __restrict__ x, bf16_t* __restrict__ xb) {
    int idx = blockIdx.x * 256 + threadIdx.x;
    if (idx >= B_ * 256 * H_) return;
    int row = idx >> 8, c = idx & 255;
    float nl = proj[(size_t)row * OUT_ + c];
    float gt = proj[(size_t)row * OUT_ + H_ + c];
    float g  = 1.f / (1.f + __expf(-gt));
    float xn = g * x[idx] + (1.f - g) * fmaxf(nl, 0.f);
    x[idx]  = xn;
    xb[idx] = f2bf(xn);
}

// ---------------------------------------------------------------------------
extern "C" void kernel_launch(void* const* d_in, const int* in_sizes, int n_in,
                              void* d_out, int out_size, void* d_ws, size_t ws_size,
                              hipStream_t stream) {
    const float* probs  = (const float*)d_in[0];
    const float* emb    = (const float*)d_in[1];
    const float* conv_w = (const float*)d_in[2];
    const float* hw_w1  = (const float*)d_in[3];
    const float* hw_b1  = (const float*)d_in[4];
    const float* hw_w2  = (const float*)d_in[5];
    const float* hw_b2  = (const float*)d_in[6];
    const float* proj_w = (const float*)d_in[7];
    const float* proj_b = (const float*)d_in[8];
    // d_in[9] = max_nb_blocks on device; recover J from out_size instead
    // (device reads are not possible under graph capture).
    long long J;
    long long rem = (long long)out_size - (long long)B_ * 256 * OUT_;
    const long long perJ = (long long)B_ * (2LL * S_ + 1);   // mask + bbp + reg
    if (rem > 0 && rem % perJ == 0 && rem / perJ >= 256) J = rem / perJ;
    else J = (long long)out_size / ((long long)B_ * (256LL * OUT_ / 256 /*per-j out*/ + 0) + perJ) > 0
             ? (long long)out_size / ((long long)B_ * (OUT_ + 2LL * S_ + 1)) : 1;
    if (J < 1) J = 1;
    int pad = (int)(J < 256 ? J : 256);

    // ---- workspace carve (all 256B aligned) ----
    char* ws = (char*)d_ws;
    auto carve = [&](size_t bytes) { char* p = ws; ws += (bytes + 255) & ~(size_t)255; return p; };
    float*  mean   = (float*) carve((size_t)B_ * S_ * 4);
    float*  stdv   = (float*) carve((size_t)B_ * S_ * 4);
    float*  invd   = (float*) carve((size_t)B_ * S_ * 4);
    bf16_t* convT  = (bf16_t*)carve((size_t)B_ * H_ * S_ * 2);
    bf16_t* Abf    = (bf16_t*)carve((size_t)B_ * 256 * S_ * 2);
    float*  xf     = (float*) carve((size_t)B_ * 256 * H_ * 4);
    bf16_t* xb     = (bf16_t*)carve((size_t)B_ * 256 * H_ * 2);
    float*  projbuf= (float*) carve((size_t)B_ * 256 * OUT_ * 4);
    bf16_t* W1t    = (bf16_t*)carve((size_t)OUT_ * H_ * 2);
    bf16_t* W2t    = (bf16_t*)carve((size_t)OUT_ * H_ * 2);
    bf16_t* PWt    = (bf16_t*)carve((size_t)OUT_ * H_ * 2);
    (void)ws_size; (void)n_in; (void)in_sizes;

    // ---- output carve ----
    float* out_main = (float*)d_out;                               // B*pad*OUT
    float* mask_o   = out_main + (size_t)B_ * pad * OUT_;          // B*J*S
    float* reg_o    = mask_o   + (size_t)B_ * J * S_;              // B*J
    float* bbp_o    = reg_o    + (size_t)B_ * J;                   // B*J*S

    // 1. prefix sums -> mean/std
    scan_kernel<<<B_, 256, 0, stream>>>(probs, mean, stdv);
    // 2. softmax denominators
    denom_kernel<<<(B_ * S_ + 255) / 256, 256, 0, stream>>>(mean, stdv, invd, (int)J);
    // 3. conv + transpose to bf16 [b][c][t]
    conv_kernel<<<dim3(B_ * (S_ / 32), H_ / 32), dim3(32, 8), 0, stream>>>(emb, conv_w, convT);
    // 4. zero GEMM-A (rows j>=J stay zero), then fused bbp/mask/token-size
    hipMemsetAsync(Abf, 0, (size_t)B_ * 256 * S_ * 2, stream);
    bbp_kernel<<<(unsigned)(B_ * J), 256, 0, stream>>>(mean, stdv, invd,
                                                       bbp_o, mask_o, reg_o, Abf, (int)J);
    // 5. weight transposes (bf16)
    wt_kernel<<<(OUT_ * H_ + 255) / 256, 256, 0, stream>>>(hw_w1, W1t, H_, OUT_);
    wt_kernel<<<(OUT_ * H_ + 255) / 256, 256, 0, stream>>>(hw_w2, W2t, H_, OUT_);
    wt_kernel<<<(OUT_ * H_ + 255) / 256, 256, 0, stream>>>(proj_w, PWt, H_, OUT_);
    // 6. block embeddings: per batch (256x2048)@(2048x256) -> xf (+bf16 mirror)
    gemm_bf16_kernel<<<dim3(32, B_), 256, 0, stream>>>(
        Abf, convT, xf, xb, nullptr, 256, H_, S_,
        (long long)256 * S_, (long long)H_ * S_, (long long)256 * H_);
    // 7. highway layer 1: (1024x256)@(256x512)+b -> proj; elementwise
    gemm_bf16_kernel<<<dim3(256, 1), 256, 0, stream>>>(
        xb, W1t, projbuf, nullptr, hw_b1, B_ * 256, OUT_, H_, 0, 0, 0);
    hw_elem_kernel<<<(B_ * 256 * H_ + 255) / 256, 256, 0, stream>>>(projbuf, xf, xb);
    // 8. highway layer 2
    gemm_bf16_kernel<<<dim3(256, 1), 256, 0, stream>>>(
        xb, W2t, projbuf, nullptr, hw_b2, B_ * 256, OUT_, H_, 0, 0, 0);
    hw_elem_kernel<<<(B_ * 256 * H_ + 255) / 256, 256, 0, stream>>>(projbuf, xf, xb);
    // 9. output projection with pad-aware packed store
    gemm_out_kernel<<<dim3(256, 1), 256, 0, stream>>>(
        xb, PWt, out_main, proj_b, B_ * 256, OUT_, H_, pad);
}
